// GRUTrajectoryDecoder_90692529422865
// MI455X (gfx1250) — compile-verified, matrix operand order checked by reading, not run on hardware
//
#include <hip/hip_runtime.h>
#include <hip/hip_bf16.h>
#include <math.h>

// ---------------------------------------------------------------------------
// GRU trajectory decoder for gfx1250 (MI455X).
//  - weights: bf16, pre-transposed [N][K] (contiguous K per output column)
//  - hidden state: bf16 resident (WMMA A-operands load directly, no cvt)
//  - GRU cell: v_wmma_f32_16x16x32_bf16, fp32 accumulate, 2 M-tiles/wave so
//    every weight (B) fragment feeds 2 WMMAs -> 2x weight-bandwidth reuse
//  - gates in fp32 VALU on the documented C/D lane layout
// ---------------------------------------------------------------------------

typedef __attribute__((ext_vector_type(16))) __bf16 v16bf;
typedef __attribute__((ext_vector_type(8)))  float  v8f;

constexpr int kB    = 512;   // batch
constexpr int kZ    = 512;   // latent dim
constexpr int kH    = 1024;  // hidden
constexpr int kG    = 3 * kH;
constexpr int kL    = 5;
constexpr int kItem = 4;
constexpr int kT    = 30;

__device__ __forceinline__ float sigmoidf_(float x) {
    return 1.0f / (1.0f + __expf(-x));
}

// ---------------------------------------------------------------------------
// Prep: convert W[K][N] fp32 -> Wt[N][K] bf16
// ---------------------------------------------------------------------------
__global__ void transpose_to_bf16(const float* __restrict__ in,
                                  __bf16* __restrict__ out, int K, int N) {
    int idx = blockIdx.x * blockDim.x + threadIdx.x;
    if (idx >= K * N) return;
    int k = idx / N, n = idx % N;
    out[(size_t)n * K + k] = (__bf16)in[idx];
}

__global__ void zero_f32(float* __restrict__ p, int n) {
    int idx = blockIdx.x * blockDim.x + threadIdx.x;
    if (idx < n) p[idx] = 0.0f;
}

// ---------------------------------------------------------------------------
// Join MLP: j = gelu_exact(z @ w1 + b1);  h0 = j @ w2 + b2 -> bf16, broadcast L
// ---------------------------------------------------------------------------
__global__ void join1_kernel(const float* __restrict__ z,
                             const float* __restrict__ w1,
                             const float* __restrict__ b1,
                             float* __restrict__ j) {
    int idx = blockIdx.x * blockDim.x + threadIdx.x;
    if (idx >= kB * 256) return;
    int b = idx >> 8, n = idx & 255;
    float s = b1[n];
    const float* zr = z + (size_t)b * kZ;
    for (int k = 0; k < kZ; ++k) s += zr[k] * w1[(size_t)k * 256 + n];
    j[idx] = 0.5f * s * (1.0f + erff(s * 0.70710678118654752f));
}

__global__ void join2_kernel(const float* __restrict__ j,
                             const float* __restrict__ w2,
                             const float* __restrict__ b2,
                             __bf16* __restrict__ hbuf0) {
    int idx = blockIdx.x * blockDim.x + threadIdx.x;
    if (idx >= kB * kH) return;
    int b = idx >> 10, n = idx & 1023;
    float s = b2[n];
    const float* jr = j + (size_t)b * 256;
    for (int k = 0; k < 256; ++k) s += jr[k] * w2[(size_t)k * kH + n];
    __bf16 v = (__bf16)s;
    for (int l = 0; l < kL; ++l)
        hbuf0[(size_t)l * kB * kH + idx] = v;   // h0 broadcast to all layers
}

// ---------------------------------------------------------------------------
// Fused GRU cell. One wave: 2 M-tiles (32 batch rows) x 16 hidden cols.
//   r = sig(gi_r+gh_r); z = sig(gi_z+gh_z); n = tanh(gi_n + r*gh_n)
//   h' = (1-z)*n + z*h
// 8 fp32 accumulators (R,Z,N,HN per M-tile). Weight fragments reused by both
// M-tiles. IN_DIM==kH -> WMMA for both GEMMs (12 WMMA/k-block);
// IN_DIM==4 (layer 0, x is fp32 step input) -> scalar seed of accumulators.
// Block = 256 threads = 8 waves: 2 wave-rows x 4 wave-cols -> 64x64 per WG.
// ---------------------------------------------------------------------------
template <int IN_DIM>
__global__ __launch_bounds__(256)
void gru_cell_wmma(const void* __restrict__ x_,        // fp32[kB,4] or bf16[kB,kH]
                   const __bf16* __restrict__ h_in,    // [kB, kH] bf16
                   __bf16* __restrict__ h_out,         // [kB, kH] bf16
                   const __bf16* __restrict__ wt_ih,   // [kG][IN_DIM] transposed
                   const __bf16* __restrict__ wt_hh,   // [kG][kH]    transposed
                   const float* __restrict__ b_ih,     // [kG]
                   const float* __restrict__ b_hh) {   // [kG]
    const int lane  = threadIdx.x & 31;
    const int wave  = threadIdx.x >> 5;
    const int row0  = (blockIdx.y * 2 + (wave >> 2)) * 32;   // 2 M-tiles
    const int col0  = (blockIdx.x * 4 + (wave & 3)) * 16;
    const int khalf = lane >> 4;
    const int l15   = lane & 15;
    const int c     = col0 + l15;        // this lane's output column (N)
    const int cr = c, cz = c + kH, cn = c + 2 * kH;

    v8f accR[2]  = {{}, {}};
    v8f accZ[2]  = {{}, {}};
    v8f accN[2]  = {{}, {}};
    v8f accHN[2] = {{}, {}};

    if (IN_DIM != kH) {
        // layer 0: K=4 input GEMM folded into accumulator init (scalar).
        // C/D layout: element v of lane -> (M = s*16 + v + khalf*8, N = l15).
        const float* x = (const float*)x_;
        #pragma unroll
        for (int s = 0; s < 2; ++s) {
            #pragma unroll
            for (int v = 0; v < 8; ++v) {
                int m = row0 + s * 16 + v + khalf * 8;
                const float* xr = x + (size_t)m * IN_DIM;
                float sr = 0.f, sz = 0.f, sn = 0.f;
                #pragma unroll
                for (int k = 0; k < IN_DIM; ++k) {
                    float xv = xr[k];
                    sr += xv * (float)wt_ih[(size_t)cr * IN_DIM + k];
                    sz += xv * (float)wt_ih[(size_t)cz * IN_DIM + k];
                    sn += xv * (float)wt_ih[(size_t)cn * IN_DIM + k];
                }
                accR[s][v] = sr; accZ[s][v] = sz; accN[s][v] = sn;
            }
        }
    }

    // A-fragments: lane holds row M = l15 (+16 for tile 1), K-half = khalf.
    const __bf16* hrow0 = h_in + (size_t)(row0 + l15) * kH;
    const __bf16* hrow1 = hrow0 + (size_t)16 * kH;
    const __bf16* xb    = (const __bf16*)x_;           // valid when IN_DIM==kH
    const __bf16* xrow0 = (IN_DIM == kH) ? (xb + (size_t)(row0 + l15) * kH) : nullptr;
    const __bf16* xrow1 = (IN_DIM == kH) ? (xrow0 + (size_t)16 * kH) : nullptr;

    const __bf16* whr = wt_hh + (size_t)cr * kH;
    const __bf16* whz = wt_hh + (size_t)cz * kH;
    const __bf16* whn = wt_hh + (size_t)cn * kH;
    const __bf16* wir = wt_ih + (size_t)cr * kH;       // only used when IN_DIM==kH
    const __bf16* wiz = wt_ih + (size_t)cz * kH;
    const __bf16* win = wt_ih + (size_t)cn * kH;

    for (int kb = 0; kb < kH / 32; ++kb) {
        const int k0 = kb * 32 + khalf * 16;

        // keep the L2 -> WGP$ weight stream ahead of the WMMA pipe
        __builtin_prefetch(whr + k0 + 32, 0, 1);
        __builtin_prefetch(whn + k0 + 32, 0, 1);

        v16bf bR = *(const v16bf*)(whr + k0);
        v16bf bZ = *(const v16bf*)(whz + k0);
        v16bf bN = *(const v16bf*)(whn + k0);
        v16bf a0 = *(const v16bf*)(hrow0 + k0);
        v16bf a1 = *(const v16bf*)(hrow1 + k0);

        accR[0]  = __builtin_amdgcn_wmma_f32_16x16x32_bf16(false, a0, false, bR,
                                                           (short)0, accR[0], false, false);
        accR[1]  = __builtin_amdgcn_wmma_f32_16x16x32_bf16(false, a1, false, bR,
                                                           (short)0, accR[1], false, false);
        accZ[0]  = __builtin_amdgcn_wmma_f32_16x16x32_bf16(false, a0, false, bZ,
                                                           (short)0, accZ[0], false, false);
        accZ[1]  = __builtin_amdgcn_wmma_f32_16x16x32_bf16(false, a1, false, bZ,
                                                           (short)0, accZ[1], false, false);
        accHN[0] = __builtin_amdgcn_wmma_f32_16x16x32_bf16(false, a0, false, bN,
                                                           (short)0, accHN[0], false, false);
        accHN[1] = __builtin_amdgcn_wmma_f32_16x16x32_bf16(false, a1, false, bN,
                                                           (short)0, accHN[1], false, false);

        if (IN_DIM == kH) {
            v16bf bRi = *(const v16bf*)(wir + k0);
            v16bf bZi = *(const v16bf*)(wiz + k0);
            v16bf bNi = *(const v16bf*)(win + k0);
            v16bf x0  = *(const v16bf*)(xrow0 + k0);
            v16bf x1  = *(const v16bf*)(xrow1 + k0);

            accR[0] = __builtin_amdgcn_wmma_f32_16x16x32_bf16(false, x0, false, bRi,
                                                              (short)0, accR[0], false, false);
            accR[1] = __builtin_amdgcn_wmma_f32_16x16x32_bf16(false, x1, false, bRi,
                                                              (short)0, accR[1], false, false);
            accZ[0] = __builtin_amdgcn_wmma_f32_16x16x32_bf16(false, x0, false, bZi,
                                                              (short)0, accZ[0], false, false);
            accZ[1] = __builtin_amdgcn_wmma_f32_16x16x32_bf16(false, x1, false, bZi,
                                                              (short)0, accZ[1], false, false);
            accN[0] = __builtin_amdgcn_wmma_f32_16x16x32_bf16(false, x0, false, bNi,
                                                              (short)0, accN[0], false, false);
            accN[1] = __builtin_amdgcn_wmma_f32_16x16x32_bf16(false, x1, false, bNi,
                                                              (short)0, accN[1], false, false);
        }
    }

    const float brs = b_ih[cr] + b_hh[cr];
    const float bzs = b_ih[cz] + b_hh[cz];
    const float bin = b_ih[cn];
    const float bhn = b_hh[cn];

    #pragma unroll
    for (int s = 0; s < 2; ++s) {
        #pragma unroll
        for (int v = 0; v < 8; ++v) {
            int m = row0 + s * 16 + v + khalf * 8;
            float r  = sigmoidf_(accR[s][v] + brs);
            float zg = sigmoidf_(accZ[s][v] + bzs);
            float n  = tanhf(accN[s][v] + bin + r * (accHN[s][v] + bhn));
            size_t o = (size_t)m * kH + c;
            float hp = (float)h_in[o];
            h_out[o] = (__bf16)((1.0f - zg) * n + zg * hp);
        }
    }
}

// ---------------------------------------------------------------------------
// Output head: pred = tanh(x @ out_w + out_b); cur = pred + step_in
// ---------------------------------------------------------------------------
__global__ void head_kernel(const __bf16* __restrict__ xlast,
                            const float* __restrict__ step_in,
                            float* __restrict__ step_out,
                            const float* __restrict__ out_w,
                            const float* __restrict__ out_b,
                            float* __restrict__ out, int t) {
    int idx = blockIdx.x * blockDim.x + threadIdx.x;
    if (idx >= kB * kItem) return;
    int b = idx >> 2, i = idx & 3;
    float s = out_b[i];
    const __bf16* xr = xlast + (size_t)b * kH;
    for (int k = 0; k < kH; ++k) s += (float)xr[k] * out_w[(size_t)k * kItem + i];
    float cur = tanhf(s) + step_in[idx];
    step_out[idx] = cur;
    out[((size_t)b * kT + t) * kItem + i] = cur;
}

// ---------------------------------------------------------------------------
extern "C" void kernel_launch(void* const* d_in, const int* in_sizes, int n_in,
                              void* d_out, int out_size, void* d_ws, size_t ws_size,
                              hipStream_t stream) {
    const float* z         = (const float*)d_in[0];
    const float* join_w1   = (const float*)d_in[1];
    const float* join_b1   = (const float*)d_in[2];
    const float* join_w2   = (const float*)d_in[3];
    const float* join_b2   = (const float*)d_in[4];
    const float* w_ih_l0   = (const float*)d_in[5];
    const float* w_hh_l0   = (const float*)d_in[6];
    const float* b_ih_l0   = (const float*)d_in[7];
    const float* b_hh_l0   = (const float*)d_in[8];
    const float* w_ih_rest = (const float*)d_in[9];
    const float* w_hh_rest = (const float*)d_in[10];
    const float* b_ih_rest = (const float*)d_in[11];
    const float* b_hh_rest = (const float*)d_in[12];
    const float* out_w     = (const float*)d_in[13];
    const float* out_b     = (const float*)d_in[14];
    (void)in_sizes; (void)n_in; (void)out_size; (void)ws_size;

    // --- workspace layout ---
    const size_t WSZ = (size_t)kG * kH;               // elems per big matrix
    char* ws = (char*)d_ws;
    size_t off = 0;
    __bf16* wt_hh  = (__bf16*)(ws + off); off += 5 * WSZ * sizeof(__bf16);
    __bf16* wt_ih  = (__bf16*)(ws + off); off += 4 * WSZ * sizeof(__bf16);
    __bf16* wt_ih0 = (__bf16*)(ws + off); off += (size_t)kG * kItem * sizeof(__bf16);
    __bf16* hbuf   = (__bf16*)(ws + off); off += 2ull * kL * kB * kH * sizeof(__bf16);
    float*  stepbf = (float*)(ws + off);  off += 2ull * kB * kItem * sizeof(float);
    float*  jbuf   = (float*)(ws + off);  off += (size_t)kB * 256 * sizeof(float);
    float*  outf   = (float*)d_out;

    // --- prep: bf16 transposed weights (L2-resident, ~57 MB) ---
    {
        dim3 tb(256), tg((unsigned)((WSZ + 255) / 256));
        transpose_to_bf16<<<tg, tb, 0, stream>>>(w_hh_l0, wt_hh, kH, kG);
        for (int i = 0; i < 4; ++i) {
            transpose_to_bf16<<<tg, tb, 0, stream>>>(w_hh_rest + (size_t)i * WSZ,
                                                     wt_hh + (size_t)(1 + i) * WSZ, kH, kG);
            transpose_to_bf16<<<tg, tb, 0, stream>>>(w_ih_rest + (size_t)i * WSZ,
                                                     wt_ih + (size_t)i * WSZ, kH, kG);
        }
        transpose_to_bf16<<<dim3((kG * kItem + 255) / 256), tb, 0, stream>>>(
            w_ih_l0, wt_ih0, kItem, kG);
    }

    // --- join MLP -> h0 (bf16) broadcast into hbuf parity 0; zero step input ---
    join1_kernel<<<dim3(kB * 256 / 256), dim3(256), 0, stream>>>(z, join_w1, join_b1, jbuf);
    join2_kernel<<<dim3(kB * kH / 256), dim3(256), 0, stream>>>(jbuf, join_w2, join_b2, hbuf);
    zero_f32<<<dim3((kB * kItem + 255) / 256), dim3(256), 0, stream>>>(stepbf, kB * kItem);

    // --- T x L rollout; double-buffered hidden state + step input ---
    const size_t HS = (size_t)kB * kH;
    dim3 gblk(256);
    dim3 ggrd(kH / 64, kB / 64);   // (16, 8) workgroups, 8 waves each

    for (int t = 0; t < kT; ++t) {
        int p = t & 1;
        __bf16* hin  = hbuf + (size_t)p * kL * HS;
        __bf16* hout = hbuf + (size_t)(1 - p) * kL * HS;
        float*  sin  = stepbf + (size_t)p * kB * kItem;
        float*  sout = stepbf + (size_t)(1 - p) * kB * kItem;

        gru_cell_wmma<kItem><<<ggrd, gblk, 0, stream>>>(
            (const void*)sin, hin, hout, wt_ih0, wt_hh, b_ih_l0, b_hh_l0);
        for (int l = 1; l < kL; ++l) {
            gru_cell_wmma<kH><<<ggrd, gblk, 0, stream>>>(
                (const void*)(hout + (size_t)(l - 1) * HS), hin + (size_t)l * HS,
                hout + (size_t)l * HS,
                wt_ih + (size_t)(l - 1) * WSZ, wt_hh + (size_t)l * WSZ,
                b_ih_rest + (size_t)(l - 1) * kG, b_hh_rest + (size_t)(l - 1) * kG);
        }
        head_kernel<<<dim3(kB * kItem / 256), gblk, 0, stream>>>(
            hout + 4 * HS, sin, sout, out_w, out_b, outf, t);
    }
}
